// Decoder_26525718020542
// MI455X (gfx1250) — matrix operand order repaired and checked by reading, not compiled, not run on hardware
//
#include <hip/hip_runtime.h>
#include <hip/hip_bf16.h>
#include <math.h>

typedef _Float16 v4h  __attribute__((ext_vector_type(4)));
typedef _Float16 v8h  __attribute__((ext_vector_type(8)));
typedef _Float16 v16h __attribute__((ext_vector_type(16)));
typedef float    v4f  __attribute__((ext_vector_type(4)));
typedef float    v8f  __attribute__((ext_vector_type(8)));

#define BM 64
#define BN 64
#define BKK 32
#define GEMM_THREADS 128

// Batched GEMM via CDNA5 WMMA f16->f32, double-buffered LDS (ping-pong).
//   TB=false: C = act(scale * A[M,K] @ B[K,N] + bias)
//   TB=true : C = act(scale * A[M,K] @ B[N,K]^T + bias)
// REQUIREMENTS (satisfied by every call in this graph):
//   M % 64 == 0, K % 32 == 0, N % 4 == 0; rows 16B-aligned.
// Batch index z = blockIdx.z decomposed as z1 = z/Z2, z2 = z%Z2 with per-matrix strides.
template<bool TB>
__global__ __launch_bounds__(GEMM_THREADS)
void gemm_wmma(const float* __restrict__ A, const float* __restrict__ Bm,
               const float* __restrict__ bias, float* __restrict__ C,
               int M, int N, int K, int ldc, float scale, int relu, int Z2,
               long sA1, long sA2, long sB1, long sB2,
               long sb1, long sb2, long sC1, long sC2)
{
    const int z  = blockIdx.z;
    const int z1 = z / Z2, z2 = z % Z2;
    A  += (long)z1 * sA1 + (long)z2 * sA2;
    Bm += (long)z1 * sB1 + (long)z2 * sB2;
    C  += (long)z1 * sC1 + (long)z2 * sC2;
    if (bias) bias += (long)z1 * sb1 + (long)z2 * sb2;

    const int tileN = blockIdx.x * BN;
    const int tileM = blockIdx.y * BM;
    const int tid  = threadIdx.x;
    const int lane = tid & 31;
    const int wv   = tid >> 5;       // wave id 0..3 (wave32)
    const int hf   = lane >> 4;      // lane half
    const int l16  = lane & 15;

    __shared__ __align__(16) _Float16 As[2][BM][BKK + 8];  // [m][k], 80B rows
    __shared__ __align__(16) _Float16 Bs[2][BN][BKK + 8];  // [n][k]

    // ---- precompute per-thread staging addresses; advance by constants per k-step
    const float* aptr[4];
    int arow[4], akol[4];
    #pragma unroll
    for (int it = 0; it < 4; ++it) {
        int e = it * 512 + tid * 4;
        arow[it] = e >> 5; akol[it] = e & 31;            // k multiple of 4
        aptr[it] = A + (long)(tileM + arow[it]) * K + akol[it];
    }
    const float* bptr[4];
    int brow[4], bkol[4];
    float bmask[4];
    #pragma unroll
    for (int it = 0; it < 4; ++it) {
        int e = it * 512 + tid * 4;
        if (TB) {
            int n = e >> 5, k = e & 31;
            int gn = tileN + n;
            brow[it] = n; bkol[it] = k;
            bmask[it] = (gn < N) ? 1.0f : 0.0f;
            bptr[it] = Bm + (long)((gn < N) ? gn : 0) * K + k;          // clamp: stay in-bounds
        } else {
            int k = e >> 6, n = e & 63;                  // vector along n (n multiple of 4)
            int gn = tileN + n;
            brow[it] = n; bkol[it] = k;
            bmask[it] = (gn < N) ? 1.0f : 0.0f;
            bptr[it] = Bm + (long)k * N + ((gn < N) ? gn : (N - 4));    // clamp
        }
    }
    const long bstep = TB ? (long)BKK : (long)BKK * N;

    v8f acc[4];
    #pragma unroll
    for (int t = 0; t < 4; ++t)
        #pragma unroll
        for (int i = 0; i < 8; ++i) acc[t][i] = 0.0f;

    const int nk = K / BKK;
    v4f av[4], bv[4];

    // ---- prologue: fetch + store tile 0 into buffer 0
    #pragma unroll
    for (int it = 0; it < 4; ++it) { av[it] = *(const v4f*)aptr[it]; aptr[it] += BKK; }
    #pragma unroll
    for (int it = 0; it < 4; ++it) { bv[it] = *(const v4f*)bptr[it]; bptr[it] += bstep; }
    #pragma unroll
    for (int it = 0; it < 4; ++it) {
        v4h h;
        #pragma unroll
        for (int j = 0; j < 4; ++j) h[j] = (_Float16)av[it][j];
        *(v4h*)(&As[0][arow[it]][akol[it]]) = h;
    }
    #pragma unroll
    for (int it = 0; it < 4; ++it) {
        v4h h;
        #pragma unroll
        for (int j = 0; j < 4; ++j) h[j] = (_Float16)(bv[it][j] * bmask[it]);
        if (TB) *(v4h*)(&Bs[0][brow[it]][bkol[it]]) = h;
        else {
            #pragma unroll
            for (int j = 0; j < 4; ++j) Bs[0][brow[it] + j][bkol[it]] = h[j];
        }
    }
    __syncthreads();

    int p = 0;
    for (int i = 0; i < nk; ++i) {
        const bool more = (i + 1 < nk);
        // prefetch next tile into registers (overlaps with WMMA below)
        if (more) {
            #pragma unroll
            for (int it = 0; it < 4; ++it) { av[it] = *(const v4f*)aptr[it]; aptr[it] += BKK; }
            #pragma unroll
            for (int it = 0; it < 4; ++it) { bv[it] = *(const v4f*)bptr[it]; bptr[it] += bstep; }
        }

        // A fragment: lanes 0-15 -> K {0..7,16..23}; lanes 16-31 -> K {8..15,24..31}
        v16h af;
        {
            const _Float16* apx = &As[p][wv * 16 + l16][0];
            v8h lo = *(const v8h*)(apx + 8 * hf);
            v8h hi = *(const v8h*)(apx + 16 + 8 * hf);
            #pragma unroll
            for (int j = 0; j < 8; ++j) { af[j] = lo[j]; af[8 + j] = hi[j]; }
        }
        #pragma unroll
        for (int t = 0; t < 4; ++t) {
            // B fragment: lane = column (n); lanes 0-15 -> K {0..7,16..23}; lanes 16-31 -> K {8..15,24..31}
            const _Float16* bpx = &Bs[p][t * 16 + l16][0];
            v8h lo = *(const v8h*)(bpx + 8 * hf);
            v8h hi = *(const v8h*)(bpx + 16 + 8 * hf);
            v16h bf;
            #pragma unroll
            for (int j = 0; j < 8; ++j) { bf[j] = lo[j]; bf[8 + j] = hi[j]; }
            acc[t] = __builtin_amdgcn_wmma_f32_16x16x32_f16(
                false, af, false, bf, (short)0, acc[t], false, false);
        }

        // store prefetched tile into the other buffer; single barrier per step
        if (more) {
            const int pn = p ^ 1;
            #pragma unroll
            for (int it = 0; it < 4; ++it) {
                v4h h;
                #pragma unroll
                for (int j = 0; j < 4; ++j) h[j] = (_Float16)av[it][j];
                *(v4h*)(&As[pn][arow[it]][akol[it]]) = h;
            }
            #pragma unroll
            for (int it = 0; it < 4; ++it) {
                v4h h;
                #pragma unroll
                for (int j = 0; j < 4; ++j) h[j] = (_Float16)(bv[it][j] * bmask[it]);
                if (TB) *(v4h*)(&Bs[pn][brow[it]][bkol[it]]) = h;
                else {
                    #pragma unroll
                    for (int j = 0; j < 4; ++j) Bs[pn][brow[it] + j][bkol[it]] = h[j];
                }
            }
            __syncthreads();
            p = pn;
        }
    }

    // epilogue: C/D layout -> m = r + 8*half, n = lane&15
    #pragma unroll
    for (int t = 0; t < 4; ++t) {
        int gn = tileN + t * 16 + l16;
        if (gn < N) {
            #pragma unroll
            for (int r = 0; r < 8; ++r) {
                int gm = tileM + wv * 16 + r + 8 * hf;
                float v = acc[t][r] * scale;
                if (bias) v += bias[gn];
                if (relu) v = fmaxf(v, 0.0f);
                C[(long)gm * ldc + gn] = v;
            }
        }
    }
}

// One wave per attention row; masked softmax in-place. Rows laid out [H][B][T][S].
__global__ __launch_bounds__(128)
void softmax_rows(float* __restrict__ P, const int* __restrict__ mask,
                  int H_, int B_, int T_, int S_, int causal)
{
    int row  = blockIdx.x * 4 + (threadIdx.x >> 5);
    int lane = threadIdx.x & 31;
    int total = H_ * B_ * T_;
    if (row >= total) return;
    int t = row % T_;
    int b = (row / T_) % B_;
    float* p = P + (long)row * S_;
    const int* mrow = mask + (long)b * S_;
    float mx = -3.0e38f;
    for (int s = lane; s < S_; s += 32) {
        bool ok = (mrow[s] != 0) && (!causal || s <= t);
        if (ok) mx = fmaxf(mx, p[s]);
    }
    for (int o = 16; o > 0; o >>= 1) mx = fmaxf(mx, __shfl_xor(mx, o, 32));
    float sum = 0.0f;
    for (int s = lane; s < S_; s += 32) {
        bool ok = (mrow[s] != 0) && (!causal || s <= t);
        float e = ok ? expf(p[s] - mx) : 0.0f;
        p[s] = e;
        sum += e;
    }
    for (int o = 16; o > 0; o >>= 1) sum += __shfl_xor(sum, o, 32);
    float inv = (sum > 0.0f) ? 1.0f / sum : 0.0f;
    for (int s = lane; s < S_; s += 32) p[s] *= inv;
}

// out = LayerNorm(x + r) * g + b ; one block per row of d elements.
__global__ __launch_bounds__(128)
void add_ln_k(const float* __restrict__ x, const float* __restrict__ r,
              const float* __restrict__ g, const float* __restrict__ be,
              float* __restrict__ out, int d)
{
    int row = blockIdx.x;
    const float* xp = x + (long)row * d;
    const float* rp = r + (long)row * d;
    float* op = out + (long)row * d;
    __shared__ float red[4];
    int tid = threadIdx.x, lane = tid & 31, wv = tid >> 5;
    float s = 0.0f;
    for (int i = tid; i < d; i += 128) s += xp[i] + rp[i];
    for (int o = 16; o > 0; o >>= 1) s += __shfl_xor(s, o, 32);
    if (lane == 0) red[wv] = s;
    __syncthreads();
    float mean = (red[0] + red[1] + red[2] + red[3]) / (float)d;
    __syncthreads();
    float v = 0.0f;
    for (int i = tid; i < d; i += 128) { float z = xp[i] + rp[i] - mean; v += z * z; }
    for (int o = 16; o > 0; o >>= 1) v += __shfl_xor(v, o, 32);
    if (lane == 0) red[wv] = v;
    __syncthreads();
    float var  = (red[0] + red[1] + red[2] + red[3]) / (float)d;
    float rinv = rsqrtf(var + 1e-5f);
    for (int i = tid; i < d; i += 128)
        op[i] = (xp[i] + rp[i] - mean) * rinv * g[i] + be[i];
}

// x[bt] = emb[token[bt]] * sqrt(d) + sinusoidal PE(t)
__global__ __launch_bounds__(128)
void embed_pos_k(const int* __restrict__ tok, const float* __restrict__ emb,
                 float* __restrict__ x, int T_, int d)
{
    int bt = blockIdx.x;
    int t  = bt % T_;
    int token = tok[bt];
    const float* ep = emb + (long)token * d;
    float* xp = x + (long)bt * d;
    const float sq = 22.62741699796952f; // sqrt(512)
    for (int k = threadIdx.x; k < d; k += 128) {
        float pe = 0.0f;
        if (t != 0) {
            float expo = (float)(2 * (k >> 1)) / (float)d;
            float ang  = (float)t * powf(10000.0f, -expo);
            pe = (k & 1) ? cosf(ang) : sinf(ang);
        }
        xp[k] = ep[k] * sq + pe;
    }
}

// [H][B][T][dv] -> [B*T][H*dv] head-concat
__global__ void reorder_heads_k(const float* __restrict__ a, float* __restrict__ o,
                                int B_, int T_, int H_, int dv)
{
    long i = (long)blockIdx.x * blockDim.x + threadIdx.x;
    long total = (long)H_ * B_ * T_ * dv;
    if (i >= total) return;
    int j = (int)(i % dv);
    long r = i / dv;
    int t = (int)(r % T_); r /= T_;
    int b = (int)(r % B_);
    int h = (int)(r / B_);
    o[((long)(b * T_ + t)) * (H_ * dv) + h * dv + j] = a[i];
}

// out[bt] = [ x[bt], ctx[b] ]  (row width 2d)
__global__ __launch_bounds__(128)
void concat_ctx_k(const float* __restrict__ x, const float* __restrict__ c,
                  float* __restrict__ o, int T_, int d)
{
    int bt = blockIdx.x;
    int b  = bt / T_;
    const float* xp = x + (long)bt * d;
    const float* cp = c + (long)b * d;
    float* op = o + (long)bt * (2 * d);
    for (int i = threadIdx.x; i < d; i += 128) { op[i] = xp[i]; op[d + i] = cp[i]; }
}

extern "C" void kernel_launch(void* const* d_in, const int* in_sizes, int n_in,
                              void* d_out, int out_size, void* d_ws, size_t ws_size,
                              hipStream_t stream)
{
    (void)in_sizes; (void)n_in; (void)out_size;
    const int Bz = 32, T = 128, S = 160, H = 8, D = 512, DK = 64, DV = 64;
    const int DF = 2048, L = 4, DC = 1024, VOC = 32000;
    const int BT = Bz * T, BS = Bz * S;

    // ---- inputs (top-level dict order; params leaves in jax-sorted order) ----
    const int*   tok    = (const int*)  d_in[0];
    const float* enc    = (const float*)d_in[1];
    const float* hist   = (const float*)d_in[2];
    const float* vis    = (const float*)d_in[3];
    const int*   imask  = (const int*)  d_in[4];
    const int*   emask  = (const int*)  d_in[5];
    const float* emb    = (const float*)d_in[6];
    const float* Wk_e   = (const float*)d_in[7];
    const float* Wk_s   = (const float*)d_in[8];
    const float* Wq_e   = (const float*)d_in[9];
    const float* Wq_s   = (const float*)d_in[10];
    const float* Wv_e   = (const float*)d_in[11];
    const float* Wv_s   = (const float*)d_in[12];
    const float* bk_e   = (const float*)d_in[13];
    const float* bk_s   = (const float*)d_in[14];
    const float* bq_e   = (const float*)d_in[15];
    const float* bq_s   = (const float*)d_in[16];
    const float* bv_e   = (const float*)d_in[17];
    const float* bv_s   = (const float*)d_in[18];
    const float* f_W0   = (const float*)d_in[19];
    const float* f_W1   = (const float*)d_in[20];
    const float* f_b0   = (const float*)d_in[21];
    const float* f_b1   = (const float*)d_in[22];
    const float* fnn_W0 = (const float*)d_in[23];
    const float* fnn_W1 = (const float*)d_in[24];
    const float* fnn_b0 = (const float*)d_in[25];
    const float* fnn_b1 = (const float*)d_in[26];
    const float* h_W0   = (const float*)d_in[27];
    const float* h_W1   = (const float*)d_in[28];
    const float* h_W2   = (const float*)d_in[29];
    const float* h_W3   = (const float*)d_in[30];
    const float* h_b0   = (const float*)d_in[31];
    const float* h_b1   = (const float*)d_in[32];
    const float* h_b2   = (const float*)d_in[33];
    const float* h_b3   = (const float*)d_in[34];
    const float* ln_b   = (const float*)d_in[35];
    const float* ln_g   = (const float*)d_in[36];
    const float* v_W0   = (const float*)d_in[37];
    const float* v_W1   = (const float*)d_in[38];
    const float* v_W2   = (const float*)d_in[39];
    const float* v_W3   = (const float*)d_in[40];
    const float* v_b0   = (const float*)d_in[41];
    const float* v_b1   = (const float*)d_in[42];
    const float* v_b2   = (const float*)d_in[43];
    const float* v_b3   = (const float*)d_in[44];
    const float* oW0    = (const float*)d_in[45];
    const float* oW1    = (const float*)d_in[46];
    const float* oW2    = (const float*)d_in[47];
    const float* ob0    = (const float*)d_in[48];
    const float* ob1    = (const float*)d_in[49];
    const float* ob2    = (const float*)d_in[50];

    // ---- workspace carve (floats) ----
    size_t off = 0;
    auto alloc = [&](size_t n) { float* r = (float*)d_ws + off; off += n; return r; };
    float* xA   = alloc((size_t)BT * D);
    float* xB   = alloc((size_t)BT * D);
    float* tmp  = alloc((size_t)BT * D);
    float* q    = alloc((size_t)H * BT * DK);
    float* kk   = alloc((size_t)H * BS * DK);
    float* vv   = alloc((size_t)H * BS * DV);
    float* lg   = alloc((size_t)H * Bz * T * S);
    float* at   = alloc((size_t)H * Bz * T * DV);
    float* cat  = alloc((size_t)BT * DC);
    float* s1   = alloc((size_t)BT * (DC / 2));
    float* s2   = alloc((size_t)BT * (DC / 4));
    float* fus  = alloc((size_t)BT * (2 * DC));
    float* fush = alloc((size_t)BT * DC);
    if (off * sizeof(float) > ws_size) return;  // insufficient scratch

    auto gemm = [&](bool tb, const float* A, const float* Bm, const float* bias, float* C,
                    int M, int N, int K, int ldc, float scale, int relu, int Z1, int Z2,
                    long sA1, long sA2, long sB1, long sB2,
                    long sb1, long sb2, long sC1, long sC2) {
        dim3 grid((N + BN - 1) / BN, (M + BM - 1) / BM, Z1 * Z2);
        if (tb)
            gemm_wmma<true><<<grid, GEMM_THREADS, 0, stream>>>(
                A, Bm, bias, C, M, N, K, ldc, scale, relu, Z2,
                sA1, sA2, sB1, sB2, sb1, sb2, sC1, sC2);
        else
            gemm_wmma<false><<<grid, GEMM_THREADS, 0, stream>>>(
                A, Bm, bias, C, M, N, K, ldc, scale, relu, Z2,
                sA1, sA2, sB1, sB2, sb1, sb2, sC1, sC2);
    };

    // ---- embedding + positional encoding ----
    embed_pos_k<<<BT, 128, 0, stream>>>(tok, emb, xA, T, D);

    float* cur = xA;
    float* nxt = xB;
    const float isq = 0.125f;  // 1/sqrt(64)

    for (int l = 0; l < L; ++l) {
        const float* lWq_s = Wq_s + (size_t)l * H * D * DK;
        const float* lWk_s = Wk_s + (size_t)l * H * D * DK;
        const float* lWv_s = Wv_s + (size_t)l * H * D * DV;
        const float* lbq_s = bq_s + (size_t)l * H * DK;
        const float* lbk_s = bk_s + (size_t)l * H * DK;
        const float* lbv_s = bv_s + (size_t)l * H * DV;
        const float* lWq_e = Wq_e + (size_t)l * H * D * DK;
        const float* lWk_e = Wk_e + (size_t)l * H * D * DK;
        const float* lWv_e = Wv_e + (size_t)l * H * D * DV;
        const float* lbq_e = bq_e + (size_t)l * H * DK;
        const float* lbk_e = bk_e + (size_t)l * H * DK;
        const float* lbv_e = bv_e + (size_t)l * H * DV;
        const float* lh_W0 = h_W0 + (size_t)l * DC * (DC / 2);
        const float* lh_W1 = h_W1 + (size_t)l * (DC / 2) * (DC / 4);
        const float* lh_W2 = h_W2 + (size_t)l * (DC / 4) * (DC / 2);
        const float* lh_W3 = h_W3 + (size_t)l * (DC / 2) * DC;
        const float* lh_b0 = h_b0 + (size_t)l * (DC / 2);
        const float* lh_b1 = h_b1 + (size_t)l * (DC / 4);
        const float* lh_b2 = h_b2 + (size_t)l * (DC / 2);
        const float* lh_b3 = h_b3 + (size_t)l * DC;
        const float* lv_W0 = v_W0 + (size_t)l * DC * (DC / 2);
        const float* lv_W1 = v_W1 + (size_t)l * (DC / 2) * (DC / 4);
        const float* lv_W2 = v_W2 + (size_t)l * (DC / 4) * (DC / 2);
        const float* lv_W3 = v_W3 + (size_t)l * (DC / 2) * DC;
        const float* lv_b0 = v_b0 + (size_t)l * (DC / 2);
        const float* lv_b1 = v_b1 + (size_t)l * (DC / 4);
        const float* lv_b2 = v_b2 + (size_t)l * (DC / 2);
        const float* lv_b3 = v_b3 + (size_t)l * DC;
        const float* lf_W0 = f_W0 + (size_t)l * (2 * DC) * DC;
        const float* lf_b0 = f_b0 + (size_t)l * DC;
        const float* lf_W1 = f_W1 + (size_t)l * DC * D;
        const float* lf_b1 = f_b1 + (size_t)l * D;
        const float* lq_W0 = fnn_W0 + (size_t)l * D * DF;
        const float* lq_b0 = fnn_b0 + (size_t)l * DF;
        const float* lq_W1 = fnn_W1 + (size_t)l * DF * D;
        const float* lq_b1 = fnn_b1 + (size_t)l * D;
        const float* lgg   = ln_g + (size_t)l * D;
        const float* lgb   = ln_b + (size_t)l * D;

        // ===== self attention =====
        gemm(false, cur, lWq_s, lbq_s, q,  BT, DK, D, DK, 1.f, 0, H, 1,
             0, 0, (long)D * DK, 0, DK, 0, (long)BT * DK, 0);
        gemm(false, cur, lWk_s, lbk_s, kk, BT, DK, D, DK, 1.f, 0, H, 1,
             0, 0, (long)D * DK, 0, DK, 0, (long)BT * DK, 0);
        gemm(false, cur, lWv_s, lbv_s, vv, BT, DV, D, DV, 1.f, 0, H, 1,
             0, 0, (long)D * DV, 0, DV, 0, (long)BT * DV, 0);
        gemm(true, q, kk, nullptr, lg, T, T, DK, T, isq, 0, H, Bz,
             (long)BT * DK, (long)T * DK, (long)BT * DK, (long)T * DK,
             0, 0, (long)Bz * T * T, (long)T * T);
        softmax_rows<<<(H * Bz * T + 3) / 4, 128, 0, stream>>>(lg, imask, H, Bz, T, T, 1);
        gemm(false, lg, vv, nullptr, at, T, DV, T, DV, 1.f, 0, H, Bz,
             (long)Bz * T * T, (long)T * T, (long)BT * DV, (long)T * DV,
             0, 0, (long)Bz * T * DV, (long)T * DV);
        reorder_heads_k<<<(H * Bz * T * DV + 255) / 256, 256, 0, stream>>>(at, tmp, Bz, T, H, DV);
        add_ln_k<<<BT, 128, 0, stream>>>(cur, tmp, lgg, lgb, nxt, D);
        { float* sw = cur; cur = nxt; nxt = sw; }   // cur = x1

        // ===== cross attention (encoder) =====
        gemm(false, cur, lWq_e, lbq_e, q,  BT, DK, D, DK, 1.f, 0, H, 1,
             0, 0, (long)D * DK, 0, DK, 0, (long)BT * DK, 0);
        gemm(false, enc, lWk_e, lbk_e, kk, BS, DK, D, DK, 1.f, 0, H, 1,
             0, 0, (long)D * DK, 0, DK, 0, (long)BS * DK, 0);
        gemm(false, enc, lWv_e, lbv_e, vv, BS, DV, D, DV, 1.f, 0, H, 1,
             0, 0, (long)D * DV, 0, DV, 0, (long)BS * DV, 0);
        gemm(true, q, kk, nullptr, lg, T, S, DK, S, isq, 0, H, Bz,
             (long)BT * DK, (long)T * DK, (long)BS * DK, (long)S * DK,
             0, 0, (long)Bz * T * S, (long)T * S);
        softmax_rows<<<(H * Bz * T + 3) / 4, 128, 0, stream>>>(lg, emask, H, Bz, T, S, 0);
        gemm(false, lg, vv, nullptr, at, T, DV, S, DV, 1.f, 0, H, Bz,
             (long)Bz * T * S, (long)T * S, (long)BS * DV, (long)S * DV,
             0, 0, (long)Bz * T * DV, (long)T * DV);
        reorder_heads_k<<<(H * Bz * T * DV + 255) / 256, 256, 0, stream>>>(at, tmp, Bz, T, H, DV);
        add_ln_k<<<BT, 128, 0, stream>>>(cur, tmp, lgg, lgb, nxt, D);
        { float* sw = cur; cur = nxt; nxt = sw; }   // cur = x2

        // ===== context streams (faithful swap: hist-concat -> v_W*, vis-concat -> h_W*) =====
        concat_ctx_k<<<BT, 128, 0, stream>>>(cur, hist, cat, T, D);
        gemm(false, cat, lv_W0, lv_b0, s1, BT, DC / 2, DC, DC / 2, 1.f, 0, 1, 1, 0,0,0,0,0,0,0,0);
        gemm(false, s1,  lv_W1, lv_b1, s2, BT, DC / 4, DC / 2, DC / 4, 1.f, 1, 1, 1, 0,0,0,0,0,0,0,0);
        gemm(false, s2,  lv_W2, lv_b2, s1, BT, DC / 2, DC / 4, DC / 2, 1.f, 0, 1, 1, 0,0,0,0,0,0,0,0);
        gemm(false, s1,  lv_W3, lv_b3, fus, BT, DC, DC / 2, 2 * DC, 1.f, 0, 1, 1, 0,0,0,0,0,0,0,0); // h_out
        concat_ctx_k<<<BT, 128, 0, stream>>>(cur, vis, cat, T, D);
        gemm(false, cat, lh_W0, lh_b0, s1, BT, DC / 2, DC, DC / 2, 1.f, 0, 1, 1, 0,0,0,0,0,0,0,0);
        gemm(false, s1,  lh_W1, lh_b1, s2, BT, DC / 4, DC / 2, DC / 4, 1.f, 1, 1, 1, 0,0,0,0,0,0,0,0);
        gemm(false, s2,  lh_W2, lh_b2, s1, BT, DC / 2, DC / 4, DC / 2, 1.f, 0, 1, 1, 0,0,0,0,0,0,0,0);
        gemm(false, s1,  lh_W3, lh_b3, fus + DC, BT, DC, DC / 2, 2 * DC, 1.f, 0, 1, 1, 0,0,0,0,0,0,0,0); // v_out
        gemm(false, fus,  lf_W0, lf_b0, fush, BT, DC, 2 * DC, DC, 1.f, 1, 1, 1, 0,0,0,0,0,0,0,0);
        gemm(false, fush, lf_W1, lf_b1, tmp,  BT, D,  DC,     D,  1.f, 0, 1, 1, 0,0,0,0,0,0,0,0);
        add_ln_k<<<BT, 128, 0, stream>>>(cur, tmp, lgg, lgb, nxt, D);
        { float* sw = cur; cur = nxt; nxt = sw; }   // cur = x3

        // ===== feed-forward =====
        gemm(false, cur, lq_W0, lq_b0, fus, BT, DF, D, DF, 1.f, 1, 1, 1, 0,0,0,0,0,0,0,0);
        gemm(false, fus, lq_W1, lq_b1, tmp, BT, D, DF, D, 1.f, 0, 1, 1, 0,0,0,0,0,0,0,0);
        add_ln_k<<<BT, 128, 0, stream>>>(cur, tmp, lgg, lgb, nxt, D);
        { float* sw = cur; cur = nxt; nxt = sw; }   // cur = x4
    }

    // ===== output head =====
    gemm(false, cur, oW0, ob0, s2, BT, D / 2, D, D / 2, 1.f, 1, 1, 1, 0,0,0,0,0,0,0,0);
    gemm(false, s2,  oW1, ob1, s1, BT, D / 4, D / 2, D / 4, 1.f, 1, 1, 1, 0,0,0,0,0,0,0,0);
    gemm(false, s1,  oW2, ob2, (float*)d_out, BT, VOC, D / 4, VOC, 1.f, 0, 1, 1, 0,0,0,0,0,0,0,0);
}
